// DeformableConv2d_85581518340270
// MI455X (gfx1250) — matrix operand order repaired and checked by reading, not compiled, not run on hardware
//
#include <hip/hip_runtime.h>

// ---------------------------------------------------------------------------
// DeformableConv2d pipeline for MI455X (gfx1250, wave32, WMMA f16 -> f32 acc)
//   k0: xh = x (NCHW f32) -> NHWC f16 with zeroed 1-px halo; zero yh halo
//   k1: off = conv3x3(xh, w_off)+b_off -> pixel-major fp32, stride 20 (ws)
//   k2: yh  = deform_conv3x3(xh, off, w_conv)+b_conv -> NHWC f16 halo buf (ws)
//   k3: out = conv3x3(yh, w_conv)+b_conv -> NCHW fp32 (d_out)
// Implicit GEMM on v_wmma_f32_16x16x32_f16, K ordered k = tap*64 + c so each
// 32-wide K-step covers one tap / 32 consecutive channels. NHWC f16 staging
// with a zero halo means: fragment = one b128-pair load at a compile-time
// immediate tap offset, no boundary compares/selects; deform bilinear lerp
// runs in packed f16 (v_pk_fma_f16) on contiguous corner vectors.
//   A (16x32) = weight fragments (M = output channels), staged in LDS as f16
//   B (32x16) = im2col / bilinear-sampled patches (N = 16 consecutive pixels)
//   D (16x16) = fp32 accumulators
// ---------------------------------------------------------------------------

typedef __attribute__((ext_vector_type(16))) _Float16 v16h;
typedef __attribute__((ext_vector_type(8)))  _Float16 v8h;
typedef __attribute__((ext_vector_type(8)))  float    v8f;

#define B_    4
#define C_    64
#define H_    192
#define W_    192
#define O_    64
#define K2_   9
#define HW_   (H_ * W_)
#define PW_   (W_ + 2)            // padded width  (zero halo)
#define PH_   (H_ + 2)            // padded height (zero halo)
#define PHW_  (PH_ * PW_)
#define CK_   (C_ * K2_)          // 576 = GEMM K
#define KT_   (CK_ / 32)          // 18 WMMA K-steps
#define TPR_  (W_ / 16)           // 12 pixel-tiles per row
#define NTILES_ (B_ * H_ * TPR_)  // 9216 pixel tiles (GEMM N)
#define WAVES_ 8                  // waves per workgroup (256 threads)
#define NOFF_  18                 // offset channels (2*K*K)
#define OFFP_  20                 // padded per-pixel offset stride (floats)

// A-matrix (16x32, f16) fragment K index for element j of lane L (contiguous: k = 16*(L/16) + j, matches B)
__device__ __forceinline__ int frag_kA(int lane, int j) {
    return ((lane >> 4) << 4) + j;
}

__device__ __forceinline__ v8f wmma16(v16h a, v16h b, v8f c) {
    return __builtin_amdgcn_wmma_f32_16x16x32_f16(
        /*neg_a=*/false, a, /*neg_b=*/false, b,
        /*c_mod=*/(short)0, c, /*reuse_a=*/false, /*reuse_b=*/false);
}

// Precomputed bilinear sample: 4 padded-pixel offsets + masked f32 weights
struct Samp2 {
    int   o00, o01, o10, o11;     // padded pixel index (iy+1)*PW + (ix+1)
    float w00, w01, w10, w11;     // validity-masked bilinear weights
};

__device__ __forceinline__ int iclamp(int v, int lo, int hi) {
    return v < lo ? lo : (v > hi ? hi : v);
}

// ---------------------------------------------------------------------------
// Kernel 0: x NCHW f32 -> padded NHWC f16 (interior); zero halos of xh AND yh
// ---------------------------------------------------------------------------
__global__ void k_stage_nhwc(const float* __restrict__ x,
                             _Float16* __restrict__ xh,
                             _Float16* __restrict__ yh) {
    int pp = blockIdx.x * blockDim.x + threadIdx.x;   // 0 .. B*PHW-1
    if (pp >= B_ * PHW_) return;
    int b  = pp / PHW_;
    int s  = pp - b * PHW_;
    int py = s / PW_;
    int pxx = s - py * PW_;
    int y = py - 1, xw = pxx - 1;

    _Float16* xop = xh + (size_t)pp * C_;
    if (((unsigned)y < H_) && ((unsigned)xw < W_)) {
        const float* xp = x + (size_t)b * C_ * HW_ + y * W_ + xw;
#pragma unroll
        for (int c0 = 0; c0 < C_; c0 += 8) {
            v8h pk;
#pragma unroll
            for (int c = 0; c < 8; ++c)
                pk[c] = (_Float16)xp[(size_t)(c0 + c) * HW_];
            *(v8h*)(xop + c0) = pk;
        }
    } else {
        const v8h z = {};
        _Float16* yop = yh + (size_t)pp * C_;
#pragma unroll
        for (int c0 = 0; c0 < C_; c0 += 8) {
            *(v8h*)(xop + c0) = z;
            *(v8h*)(yop + c0) = z;
        }
    }
}

// ---------------------------------------------------------------------------
// Kernel 1: offset conv (Cout = 18, padded to 32 -> 2 channel tiles)
// ---------------------------------------------------------------------------
__global__ void k_offset_conv(const _Float16* __restrict__ xh,
                              const float* __restrict__ w_off,
                              const float* __restrict__ b_off,
                              float* __restrict__ offp) {
    extern __shared__ char smem_raw[];
    _Float16* wlds = (_Float16*)smem_raw;   // [KT_][2][32][16]

    const int tid  = threadIdx.x;
    const int lane = tid & 31;
    const int wave = tid >> 5;

    const int WTOT = KT_ * 2 * 32 * 16;
    for (int f = tid; f < WTOT; f += blockDim.x) {
        int j  = f & 15;
        int l  = (f >> 4) & 31;
        int mt = (f >> 9) & 1;
        int kt = f >> 10;
        int ch = mt * 16 + (l & 15);
        int k  = kt * 32 + frag_kA(l, j);
        int c  = k & 63;
        int t  = k >> 6;
        float v = (ch < NOFF_) ? w_off[ch * CK_ + c * K2_ + t] : 0.0f;
        wlds[f] = (_Float16)v;
    }
    __syncthreads();

    const int tile = blockIdx.x * WAVES_ + wave;
    const int b  = tile / (H_ * TPR_);
    const int r  = tile % (H_ * TPR_);
    const int h  = r / TPR_;
    const int w0 = (r % TPR_) * 16;
    const int px = w0 + (lane & 15);
    const int cb = (lane >> 4) << 4;

    // lane's tap(0,0) position in the padded image: row h, col px (halo shift)
    const _Float16* xrow =
        xh + ((size_t)b * PHW_ + (size_t)h * PW_ + px) * C_ + cb;

    v8f acc0 = {};
    v8f acc1 = {};
#pragma unroll
    for (int t = 0; t < 9; ++t) {
        const int ky = t / 3, kx = t % 3;            // compile-time
        const _Float16* xt = xrow + (ky * PW_ + kx) * C_;  // immediate offset
#pragma unroll
        for (int half = 0; half < 2; ++half) {
            v16h bf = *(const v16h*)(xt + half * 32);
            const int kt = t * 2 + half;
            v16h a0 = *(const v16h*)&wlds[(kt * 2 + 0) * 512 + lane * 16];
            v16h a1 = *(const v16h*)&wlds[(kt * 2 + 1) * 512 + lane * 16];
            acc0 = wmma16(a0, bf, acc0);
            acc1 = wmma16(a1, bf, acc1);
        }
    }

    const int mrow = (lane >> 4) << 3;        // D row base for this half-wave
    float* op = offp + (size_t)((b * H_ + h) * W_ + px) * OFFP_;
#pragma unroll
    for (int v = 0; v < 8; ++v)
        op[mrow + v] = acc0[v] + b_off[mrow + v];   // ch 0..15
    if (mrow == 0) {                                 // ch 16,17 live in acc1
        op[16] = acc1[0] + b_off[16];
        op[17] = acc1[1] + b_off[17];
    }
}

// ---------------------------------------------------------------------------
// Kernel 2: deformable conv (Cout = 64 -> 4 channel tiles), y padded NHWC f16
// ---------------------------------------------------------------------------
__global__ void k_deform_conv(const _Float16* __restrict__ xh,
                              const float* __restrict__ offp,
                              const float* __restrict__ w_conv,
                              const float* __restrict__ b_conv,
                              _Float16* __restrict__ yh) {
    extern __shared__ char smem_raw[];
    _Float16* wlds = (_Float16*)smem_raw;                       // KT_*4*512 f16
    Samp2* samp_all = (Samp2*)(smem_raw + KT_ * 4 * 512 * 2);   // 8 waves x [9][16]

    const int tid  = threadIdx.x;
    const int lane = tid & 31;
    const int wave = tid >> 5;

    const int WTOT = KT_ * 4 * 32 * 16;
    for (int f = tid; f < WTOT; f += blockDim.x) {
        int j  = f & 15;
        int l  = (f >> 4) & 31;
        int mt = (f >> 9) & 3;
        int kt = f >> 11;
        int ch = mt * 16 + (l & 15);
        int k  = kt * 32 + frag_kA(l, j);
        int c  = k & 63;
        int t  = k >> 6;
        wlds[f] = (_Float16)w_conv[ch * CK_ + c * K2_ + t];
    }

    const int tile = blockIdx.x * WAVES_ + wave;
    const int b  = tile / (H_ * TPR_);
    const int r  = tile % (H_ * TPR_);
    const int h  = r / TPR_;
    const int w0 = (r % TPR_) * 16;
    const int p  = lane & 15;
    const int cb = (lane >> 4) << 4;

    // Per-wave bilinear params: layout [tap][pixel]
    Samp2* samp = samp_all + wave * (9 * 16);
    for (int e = lane; e < 9 * 16; e += 32) {
        int pp = e & 15;
        int t  = e >> 4;
        int ky = t / 3, kx = t - 3 * (t / 3);
        int w  = w0 + pp;
        const float* op = offp + (size_t)((b * H_ + h) * W_ + w) * OFFP_;
        float dy = op[2 * t + 0];
        float dx = op[2 * t + 1];
        float pyf = dy + (float)(ky + h - 1);
        float pxf = dx + (float)(kx + w - 1);
        float fy0 = floorf(pyf);
        float fx0 = floorf(pxf);
        int iy = (int)fy0, ix = (int)fx0;
        float ay = pyf - fy0, ax = pxf - fx0;
        bool y0v = (unsigned)iy < H_,       y1v = (unsigned)(iy + 1) < H_;
        bool x0v = (unsigned)ix < W_,       x1v = (unsigned)(ix + 1) < W_;
        int iy0 = iclamp(iy,     0, H_ - 1), iy1 = iclamp(iy + 1, 0, H_ - 1);
        int ix0 = iclamp(ix,     0, W_ - 1), ix1 = iclamp(ix + 1, 0, W_ - 1);
        Samp2 s;   // padded-image pixel indices
        s.o00 = (iy0 + 1) * PW_ + (ix0 + 1);  s.o01 = (iy0 + 1) * PW_ + (ix1 + 1);
        s.o10 = (iy1 + 1) * PW_ + (ix0 + 1);  s.o11 = (iy1 + 1) * PW_ + (ix1 + 1);
        s.w00 = (y0v && x0v) ? (1.0f - ay) * (1.0f - ax) : 0.0f;
        s.w01 = (y0v && x1v) ? (1.0f - ay) * ax          : 0.0f;
        s.w10 = (y1v && x0v) ? ay * (1.0f - ax)          : 0.0f;
        s.w11 = (y1v && x1v) ? ay * ax                   : 0.0f;
        samp[e] = s;
    }
    __syncthreads();

    const _Float16* xpb = xh + (size_t)b * PHW_ * C_ + cb;
    v8f acc[4] = {};

    for (int t = 0; t < 9; ++t) {
        Samp2 s = samp[t * 16 + p];           // one LDS read per tap
        _Float16 hw00 = (_Float16)s.w00;
        _Float16 hw01 = (_Float16)s.w01;
        _Float16 hw10 = (_Float16)s.w10;
        _Float16 hw11 = (_Float16)s.w11;
        size_t a00 = (size_t)s.o00 * C_;
        size_t a01 = (size_t)s.o01 * C_;
        size_t a10 = (size_t)s.o10 * C_;
        size_t a11 = (size_t)s.o11 * C_;
#pragma unroll
        for (int half = 0; half < 2; ++half) {
            const _Float16* xc = xpb + half * 32;
            v16h x00 = *(const v16h*)(xc + a00);
            v16h x01 = *(const v16h*)(xc + a01);
            v16h x10 = *(const v16h*)(xc + a10);
            v16h x11 = *(const v16h*)(xc + a11);
            v16h bf = x00 * hw00 + x01 * hw01 + x10 * hw10 + x11 * hw11;
            const int kt = t * 2 + half;
#pragma unroll
            for (int mt = 0; mt < 4; ++mt) {
                v16h af = *(const v16h*)&wlds[(kt * 4 + mt) * 512 + lane * 16];
                acc[mt] = wmma16(af, bf, acc[mt]);
            }
        }
    }

    const int mrow = (lane >> 4) << 3;
    _Float16* yp = yh + ((size_t)b * PHW_ + (size_t)(h + 1) * PW_ + (w0 + p + 1)) * O_;
#pragma unroll
    for (int mt = 0; mt < 4; ++mt) {
        const int ch0 = mt * 16 + mrow;
        v8h pk;
#pragma unroll
        for (int v = 0; v < 8; ++v)
            pk[v] = (_Float16)(acc[mt][v] + b_conv[ch0 + v]);
        *(v8h*)(yp + ch0) = pk;
    }
}

// ---------------------------------------------------------------------------
// Kernel 3: final conv on padded NHWC f16 y (4 tiles), NCHW fp32 out
// ---------------------------------------------------------------------------
__global__ void k_out_conv(const _Float16* __restrict__ yh,
                           const float* __restrict__ w_conv,
                           const float* __restrict__ b_conv,
                           float* __restrict__ out) {
    extern __shared__ char smem_raw[];
    _Float16* wlds = (_Float16*)smem_raw;   // KT_*4*512 f16

    const int tid  = threadIdx.x;
    const int lane = tid & 31;
    const int wave = tid >> 5;

    const int WTOT = KT_ * 4 * 32 * 16;
    for (int f = tid; f < WTOT; f += blockDim.x) {
        int j  = f & 15;
        int l  = (f >> 4) & 31;
        int mt = (f >> 9) & 3;
        int kt = f >> 11;
        int ch = mt * 16 + (l & 15);
        int k  = kt * 32 + frag_kA(l, j);
        int c  = k & 63;
        int t  = k >> 6;
        wlds[f] = (_Float16)w_conv[ch * CK_ + c * K2_ + t];
    }
    __syncthreads();

    const int tile = blockIdx.x * WAVES_ + wave;
    const int b  = tile / (H_ * TPR_);
    const int r  = tile % (H_ * TPR_);
    const int h  = r / TPR_;
    const int w0 = (r % TPR_) * 16;
    const int px = w0 + (lane & 15);
    const int cb = (lane >> 4) << 4;

    const _Float16* yrow =
        yh + ((size_t)b * PHW_ + (size_t)h * PW_ + px) * C_ + cb;

    v8f acc[4] = {};
#pragma unroll
    for (int t = 0; t < 9; ++t) {
        const int ky = t / 3, kx = t % 3;            // compile-time
        const _Float16* yt = yrow + (ky * PW_ + kx) * C_;  // immediate offset
#pragma unroll
        for (int half = 0; half < 2; ++half) {
            v16h bf = *(const v16h*)(yt + half * 32);
            const int kt = t * 2 + half;
#pragma unroll
            for (int mt = 0; mt < 4; ++mt) {
                v16h af = *(const v16h*)&wlds[(kt * 4 + mt) * 512 + lane * 16];
                acc[mt] = wmma16(af, bf, acc[mt]);
            }
        }
    }

    const int mrow = (lane >> 4) << 3;
#pragma unroll
    for (int mt = 0; mt < 4; ++mt) {
#pragma unroll
        for (int v = 0; v < 8; ++v) {
            int ch = mt * 16 + mrow + v;
            out[((b * O_ + ch) * H_ + h) * W_ + px] = acc[mt][v] + b_conv[ch];
        }
    }
}

// ---------------------------------------------------------------------------
extern "C" void kernel_launch(void* const* d_in, const int* in_sizes, int n_in,
                              void* d_out, int out_size, void* d_ws, size_t ws_size,
                              hipStream_t stream) {
    const float* x      = (const float*)d_in[0];
    const float* w_off  = (const float*)d_in[1];
    const float* b_off  = (const float*)d_in[2];
    const float* w_conv = (const float*)d_in[3];
    const float* b_conv = (const float*)d_in[4];
    float* out = (float*)d_out;

    // workspace: [xh padded NHWC f16][offp f32 pad20][yh padded NHWC f16]
    char* ws = (char*)d_ws;
    _Float16* xhN = (_Float16*)ws;                                   // 19.3 MB
    size_t xh_bytes  = (size_t)B_ * PHW_ * C_ * sizeof(_Float16);
    float* offp = (float*)(ws + xh_bytes);                           // 11.8 MB
    size_t off_bytes = (size_t)B_ * HW_ * OFFP_ * sizeof(float);
    _Float16* yhN = (_Float16*)(ws + xh_bytes + off_bytes);          // 19.3 MB

    dim3 grid(NTILES_ / WAVES_);   // 1152 workgroups
    dim3 block(WAVES_ * 32);       // 256 threads = 8 wave32

    size_t lds1 = (size_t)KT_ * 2 * 512 * sizeof(_Float16);              // 36 KB
    size_t lds2 = (size_t)KT_ * 4 * 512 * sizeof(_Float16)
                + (size_t)WAVES_ * 9 * 16 * sizeof(Samp2);               // 108 KB
    size_t lds3 = (size_t)KT_ * 4 * 512 * sizeof(_Float16);              // 72 KB

    int npad = B_ * PHW_;
    k_stage_nhwc<<<dim3((npad + 255) / 256), dim3(256), 0, stream>>>(x, xhN, yhN);
    k_offset_conv<<<grid, block, lds1, stream>>>(xhN, w_off, b_off, offp);
    k_deform_conv<<<grid, block, lds2, stream>>>(xhN, offp, w_conv, b_conv, yhN);
    k_out_conv  <<<grid, block, lds3, stream>>>(yhN, w_conv, b_conv, out);
}
